// AtomToTokenEncoder_71949292143040
// MI455X (gfx1250) — matrix operand order repaired and checked by reading, not compiled, not run on hardware
//
#include <hip/hip_runtime.h>

typedef __attribute__((ext_vector_type(16))) _Float16 v16h;
typedef __attribute__((ext_vector_type(8)))  _Float16 v8h;
typedef __attribute__((ext_vector_type(8)))  float    v8f;

#define B_SZ     2
#define N_ATOM   2048
#define D_ATOM   128
#define H_HEADS  4
#define D_HH     32
#define D_FF     512
#define N_PAIRS  16384
#define N_TOK    512
#define D_MODEL  512
#define ROWS     (B_SZ * N_ATOM)      // 4096
#define SEG_CAP  64

// ---------------------------------------------------------------------------
// fp32 -> fp16 TRANSPOSED weight repack: src is [rows=K][cols=N] row-major,
// dst is [N_total][K] (column-concat via dstOff). Lets the GEMM read WMMA
// B fragments as two contiguous 128-bit global loads (no LDS staging).
// ---------------------------------------------------------------------------
__global__ void conv_f16t_kernel(const float* __restrict__ src, _Float16* __restrict__ dst,
                                 int rows, int cols, int ldK, int dstOff) {
    int i = blockIdx.x * blockDim.x + threadIdx.x;
    if (i >= rows * cols) return;
    int r = i / cols, c = i % cols;                 // r = k, c = n
    dst[(size_t)(dstOff + c) * ldK + r] = (_Float16)src[i];
}

__global__ void fill_u32_kernel(unsigned* __restrict__ p, unsigned v, int n) {
    int i = blockIdx.x * blockDim.x + threadIdx.x;
    if (i < n) p[i] = v;
}

// ---------------------------------------------------------------------------
// WMMA GEMM:  C[M,N] (f32 acc) = A[M,K] (f16 row-major) x Bt[N,K] (f16, B^T)
// Block = 128 threads = 4 waves; each wave computes a 16x32 D tile (two
// accumulators sharing one A fragment). Block tile = 64x32. K templated so
// the loop fully unrolls; no LDS, no barriers.
// Epilogue: + bias[N], + addF[M,N]; writes f32 (Cf) and/or f16 (Ch).
// ---------------------------------------------------------------------------
template <int K>
__global__ __launch_bounds__(128)
void wmma_gemm_kernel(const _Float16* __restrict__ A, const _Float16* __restrict__ Bt,
                      float* __restrict__ Cf, _Float16* __restrict__ Ch,
                      const float* __restrict__ addF, const float* __restrict__ bias,
                      int M, int N) {
    const int m0   = blockIdx.x * 64;
    const int n0   = blockIdx.y * 32;
    const int wave = threadIdx.x >> 5;
    const int lane = threadIdx.x & 31;
    const int g    = lane >> 4;     // half-wave group
    const int ml   = lane & 15;
    const int mrow = m0 + wave * 16 + ml;

    v8f acc0 = {0.f, 0.f, 0.f, 0.f, 0.f, 0.f, 0.f, 0.f};
    v8f acc1 = {0.f, 0.f, 0.f, 0.f, 0.f, 0.f, 0.f, 0.f};

    const _Float16* arow  = A  + (size_t)mrow * K;
    const _Float16* brow0 = Bt + (size_t)(n0 + ml) * K;        // tile 0 column
    const _Float16* brow1 = Bt + (size_t)(n0 + 16 + ml) * K;   // tile 1 column

#pragma unroll
    for (int k0 = 0; k0 < K; k0 += 32) {
        // A fragment: elem i -> K = k0 + g*16 + i (16 contiguous halves)
        const v8h* pa = (const v8h*)(arow + k0 + g * 16);
        v8h alo = pa[0];
        v8h ahi = pa[1];                    // +8 halves
        v16h af;
#pragma unroll
        for (int i = 0; i < 8; ++i) { af[i] = alo[i]; af[8 + i] = ahi[i]; }

        // B fragments: elem i -> K = k0 + g*16 + i (16 contiguous halves)
        const v8h* pb0 = (const v8h*)(brow0 + k0 + g * 16);
        v8h b0lo = pb0[0], b0hi = pb0[1];
        const v8h* pb1 = (const v8h*)(brow1 + k0 + g * 16);
        v8h b1lo = pb1[0], b1hi = pb1[1];
        v16h bf0, bf1;
#pragma unroll
        for (int i = 0; i < 8; ++i) {
            bf0[i] = b0lo[i]; bf0[8 + i] = b0hi[i];
            bf1[i] = b1lo[i]; bf1[8 + i] = b1hi[i];
        }

        acc0 = __builtin_amdgcn_wmma_f32_16x16x32_f16(false, af, false, bf0,
                                                      (short)0, acc0, false, false);
        acc1 = __builtin_amdgcn_wmma_f32_16x16x32_f16(false, af, false, bf1,
                                                      (short)0, acc1, false, false);
    }

    // Store: VGPR r holds M = g*8 + r; col = lane&15 within each 16-wide tile
#pragma unroll
    for (int r = 0; r < 8; ++r) {
        int mr = m0 + wave * 16 + g * 8 + r;
#pragma unroll
        for (int tile = 0; tile < 2; ++tile) {
            int nc = n0 + tile * 16 + ml;
            float v = (tile == 0) ? acc0[r] : acc1[r];
            if (bias) v += bias[nc];
            if (addF) v += addF[(size_t)mr * N + nc];
            if (Cf) Cf[(size_t)mr * N + nc] = v;
            if (Ch) Ch[(size_t)mr * N + nc] = (_Float16)v;
        }
    }
}

// ---------------------------------------------------------------------------
// LayerNorm (one wave32 per 128-wide row) -> f16 output
// ---------------------------------------------------------------------------
__global__ void ln1_kernel(const float* __restrict__ x, const float* __restrict__ gam,
                           const float* __restrict__ bet, _Float16* __restrict__ y, int rows) {
    int wid = threadIdx.x >> 5, lane = threadIdx.x & 31;
    int row = blockIdx.x * (blockDim.x >> 5) + wid;
    if (row >= rows) return;
    const float* xr = x + (size_t)row * D_ATOM;
    float v[4], s = 0.f;
#pragma unroll
    for (int k = 0; k < 4; ++k) { v[k] = xr[lane + 32 * k]; s += v[k]; }
#pragma unroll
    for (int off = 16; off; off >>= 1) s += __shfl_xor(s, off, 32);
    float mean = s * (1.f / 128.f);
    float var = 0.f;
#pragma unroll
    for (int k = 0; k < 4; ++k) { float d = v[k] - mean; var += d * d; }
#pragma unroll
    for (int off = 16; off; off >>= 1) var += __shfl_xor(var, off, 32);
    float inv = rsqrtf(var * (1.f / 128.f) + 1e-5f);
#pragma unroll
    for (int k = 0; k < 4; ++k) {
        int c = lane + 32 * k;
        y[(size_t)row * D_ATOM + c] = (_Float16)((v[k] - mean) * inv * gam[c] + bet[c]);
    }
}

// q = c_atom + sigmoid(G)*oproj ; h = LN(q) -> f16 ; also store q (f32 residual)
__global__ void ln2_gate_kernel(const float* __restrict__ c_atom, const float* __restrict__ QKVG,
                                const float* __restrict__ oproj, const float* __restrict__ gam,
                                const float* __restrict__ bet, float* __restrict__ qres,
                                _Float16* __restrict__ h16, int rows) {
    int wid = threadIdx.x >> 5, lane = threadIdx.x & 31;
    int row = blockIdx.x * (blockDim.x >> 5) + wid;
    if (row >= rows) return;
    float v[4], s = 0.f;
#pragma unroll
    for (int k = 0; k < 4; ++k) {
        int c = lane + 32 * k;
        float gv = QKVG[(size_t)row * 512 + 384 + c];
        float sg = 1.f / (1.f + __expf(-gv));
        float q  = c_atom[(size_t)row * D_ATOM + c] + sg * oproj[(size_t)row * D_ATOM + c];
        qres[(size_t)row * D_ATOM + c] = q;
        v[k] = q; s += q;
    }
#pragma unroll
    for (int off = 16; off; off >>= 1) s += __shfl_xor(s, off, 32);
    float mean = s * (1.f / 128.f);
    float var = 0.f;
#pragma unroll
    for (int k = 0; k < 4; ++k) { float d = v[k] - mean; var += d * d; }
#pragma unroll
    for (int off = 16; off; off >>= 1) var += __shfl_xor(var, off, 32);
    float inv = rsqrtf(var * (1.f / 128.f) + 1e-5f);
#pragma unroll
    for (int k = 0; k < 4; ++k) {
        int c = lane + 32 * k;
        h16[(size_t)row * D_ATOM + c] = (_Float16)((v[k] - mean) * inv * gam[c] + bet[c]);
    }
}

// ---------------------------------------------------------------------------
// Token-segment bounds (token_idx is sorted per batch)
// ---------------------------------------------------------------------------
__global__ void seg_bounds_kernel(const int* __restrict__ tok, int* __restrict__ segs,
                                  int* __restrict__ sege) {
    int idx = blockIdx.x * blockDim.x + threadIdx.x;
    if (idx >= ROWS) return;
    int b = idx / N_ATOM, i = idx % N_ATOM;
    const int* tb = tok + (size_t)b * N_ATOM;
    int t = tb[i];
    int s = i; while (s > 0 && tb[s - 1] == t) --s;
    int e = i + 1; while (e < N_ATOM && tb[e] == t) ++e;
    segs[idx] = s; sege[idx] = e;
}

// ---------------------------------------------------------------------------
// Pair-bias scatter with exact last-write-wins (atomicMax on pair index)
// ---------------------------------------------------------------------------
__global__ void pair_scatter1_kernel(const int* __restrict__ plmidx, const int* __restrict__ tok,
                                     const int* __restrict__ segs, int* __restrict__ winner) {
    int idx = blockIdx.x * blockDim.x + threadIdx.x;
    if (idx >= B_SZ * N_PAIRS) return;
    int b = idx / N_PAIRS, p = idx % N_PAIRS;
    int i = plmidx[(size_t)idx * 2 + 0];
    int j = plmidx[(size_t)idx * 2 + 1];
    int ri = b * N_ATOM + i, rj = b * N_ATOM + j;
    if (tok[ri] != tok[rj]) return;             // bias outside mask never matters
    int jl = j - segs[ri];
    if (jl < 0 || jl >= SEG_CAP) return;
    atomicMax(&winner[(size_t)ri * SEG_CAP + jl], p);
}

__global__ void pair_scatter2_kernel(const int* __restrict__ plmidx, const int* __restrict__ tok,
                                     const int* __restrict__ segs, const int* __restrict__ winner,
                                     const float* __restrict__ plm, const float* __restrict__ pbw,
                                     const float* __restrict__ pbb, float* __restrict__ segbias) {
    int idx = blockIdx.x * blockDim.x + threadIdx.x;
    if (idx >= B_SZ * N_PAIRS) return;
    int b = idx / N_PAIRS, p = idx % N_PAIRS;
    int i = plmidx[(size_t)idx * 2 + 0];
    int j = plmidx[(size_t)idx * 2 + 1];
    int ri = b * N_ATOM + i, rj = b * N_ATOM + j;
    if (tok[ri] != tok[rj]) return;
    int jl = j - segs[ri];
    if (jl < 0 || jl >= SEG_CAP) return;
    if (winner[(size_t)ri * SEG_CAP + jl] != p) return;
    const float* pv = plm + (size_t)idx * 16;
#pragma unroll
    for (int h = 0; h < H_HEADS; ++h) {
        float v = pbb[h];
#pragma unroll
        for (int c = 0; c < 16; ++c) v += pv[c] * pbw[c * H_HEADS + h];
        segbias[((size_t)ri * SEG_CAP + jl) * H_HEADS + h] = v;
    }
}

// ---------------------------------------------------------------------------
// Segment-local attention (mask is token equality on sorted tokens => segments).
// Thread per (row, head). fp32 exact; segments are tiny (avg ~4 atoms).
// ---------------------------------------------------------------------------
__global__ void attn_kernel(const float* __restrict__ QKVG, const float* __restrict__ segbias,
                            const int* __restrict__ segs, const int* __restrict__ sege,
                            _Float16* __restrict__ attout) {
    int flat = blockIdx.x * blockDim.x + threadIdx.x;
    if (flat >= ROWS * H_HEADS) return;
    int row = flat >> 2, h = flat & 3;
    int b = row / N_ATOM;
    int s = segs[row], e = sege[row];
    const float* qp = QKVG + (size_t)row * 512 + h * D_HH;
    float q[D_HH];
#pragma unroll
    for (int d = 0; d < D_HH; ++d) q[d] = qp[d];
    const float scale = 0.17677669529663687f;   // 1/sqrt(32)

    float mmax = -1e30f;
    for (int j = s; j < e; ++j) {
        const float* kp = QKVG + (size_t)(b * N_ATOM + j) * 512 + 128 + h * D_HH;
        float dt = 0.f;
#pragma unroll
        for (int d = 0; d < D_HH; ++d) dt += q[d] * kp[d];
        float sc = dt * scale;
        int jl = j - s;
        if (jl < SEG_CAP) sc += segbias[((size_t)row * SEG_CAP + jl) * H_HEADS + h];
        mmax = fmaxf(mmax, sc);
    }
    float sum = 0.f, o[D_HH];
#pragma unroll
    for (int d = 0; d < D_HH; ++d) o[d] = 0.f;
    for (int j = s; j < e; ++j) {
        const float* kp = QKVG + (size_t)(b * N_ATOM + j) * 512 + 128 + h * D_HH;
        float dt = 0.f;
#pragma unroll
        for (int d = 0; d < D_HH; ++d) dt += q[d] * kp[d];
        float sc = dt * scale;
        int jl = j - s;
        if (jl < SEG_CAP) sc += segbias[((size_t)row * SEG_CAP + jl) * H_HEADS + h];
        float w = __expf(sc - mmax);
        sum += w;
        const float* vp = QKVG + (size_t)(b * N_ATOM + j) * 512 + 256 + h * D_HH;
#pragma unroll
        for (int d = 0; d < D_HH; ++d) o[d] += w * vp[d];
    }
    float inv = 1.f / sum;
    _Float16* op = attout + (size_t)row * D_ATOM + h * D_HH;
#pragma unroll
    for (int d = 0; d < D_HH; ++d) op[d] = (_Float16)(o[d] * inv);
}

// silu(u1) * u2  -> f16 for FFN2 GEMM
__global__ void silu_mul_kernel(const float* __restrict__ u, _Float16* __restrict__ act16) {
    int i = blockIdx.x * blockDim.x + threadIdx.x;
    if (i >= ROWS * D_FF) return;
    int r = i / D_FF, c = i % D_FF;
    float a  = u[(size_t)r * (2 * D_FF) + c];
    float bb = u[(size_t)r * (2 * D_FF) + D_FF + c];
    float sg = a / (1.f + __expf(-a));
    act16[i] = (_Float16)(sg * bb);
}

// segment mean over tokens (binary search on sorted token_idx; no atomics)
__global__ void token_reduce_kernel(const float* __restrict__ atomfeat,
                                    const int* __restrict__ tok, float* __restrict__ out) {
    int b = blockIdx.x / N_TOK, t = blockIdx.x % N_TOK;
    const int* tb = tok + (size_t)b * N_ATOM;
    int lo = 0, hi = N_ATOM;
    while (lo < hi) { int m = (lo + hi) >> 1; if (tb[m] < t) lo = m + 1; else hi = m; }
    int st = lo;
    lo = st; hi = N_ATOM;
    while (lo < hi) { int m = (lo + hi) >> 1; if (tb[m] <= t) lo = m + 1; else hi = m; }
    int en = lo;
    float inv = 1.f / fmaxf((float)(en - st), 1.f);
    for (int c = threadIdx.x; c < D_MODEL; c += blockDim.x) {
        float s = 0.f;
        for (int r = st; r < en; ++r) s += atomfeat[(size_t)(b * N_ATOM + r) * D_MODEL + c];
        out[((size_t)(b * N_TOK + t)) * D_MODEL + c] = s * inv;
    }
}

// ---------------------------------------------------------------------------
extern "C" void kernel_launch(void* const* d_in, const int* in_sizes, int n_in,
                              void* d_out, int out_size, void* d_ws, size_t ws_size,
                              hipStream_t stream) {
    const float* c_atom   = (const float*)d_in[0];
    const float* p_lm     = (const float*)d_in[1];
    const int*   p_lm_idx = (const int*)d_in[2];
    const int*   tok      = (const int*)d_in[3];
    const float* ln_a_g   = (const float*)d_in[5];
    const float* ln_a_b   = (const float*)d_in[6];
    const float* w_q      = (const float*)d_in[7];
    const float* w_k      = (const float*)d_in[8];
    const float* w_v      = (const float*)d_in[9];
    const float* w_g      = (const float*)d_in[10];
    const float* w_o      = (const float*)d_in[11];
    const float* pb_w     = (const float*)d_in[12];
    const float* pb_b     = (const float*)d_in[13];
    const float* ln_f_g   = (const float*)d_in[14];
    const float* ln_f_b   = (const float*)d_in[15];
    const float* sw_w1    = (const float*)d_in[16];
    const float* sw_w2    = (const float*)d_in[17];
    const float* sw_w3    = (const float*)d_in[18];
    const float* tok_w    = (const float*)d_in[19];
    const float* tok_b    = (const float*)d_in[20];
    float* out = (float*)d_out;

    // ---- workspace carve-out (each chunk 256B aligned) ----
    size_t off = 0;
    auto alloc = [&](size_t bytes) {
        off = (off + 255) & ~(size_t)255;
        void* p = (char*)d_ws + off;
        off += bytes;
        return p;
    };
    _Float16* qn16    = (_Float16*)alloc((size_t)ROWS * D_ATOM * 2);
    _Float16* attout  = (_Float16*)alloc((size_t)ROWS * D_ATOM * 2);
    _Float16* h16     = (_Float16*)alloc((size_t)ROWS * D_ATOM * 2);
    _Float16* q2_16   = (_Float16*)alloc((size_t)ROWS * D_ATOM * 2);
    _Float16* act16   = (_Float16*)alloc((size_t)ROWS * D_FF * 2);
    _Float16* WqkvgT  = (_Float16*)alloc((size_t)512 * D_ATOM * 2);        // [512][128]
    _Float16* WoT     = (_Float16*)alloc((size_t)D_ATOM * D_ATOM * 2);     // [128][128]
    _Float16* W12T    = (_Float16*)alloc((size_t)(2 * D_FF) * D_ATOM * 2); // [1024][128]
    _Float16* W3T     = (_Float16*)alloc((size_t)D_ATOM * D_FF * 2);       // [128][512]
    _Float16* WtokT   = (_Float16*)alloc((size_t)D_MODEL * D_ATOM * 2);    // [512][128]
    float*    QKVG    = (float*)alloc((size_t)ROWS * 512 * 4);
    float*    oproj   = (float*)alloc((size_t)ROWS * D_ATOM * 4);
    float*    qres    = (float*)alloc((size_t)ROWS * D_ATOM * 4);
    float*    u       = (float*)alloc((size_t)ROWS * 2 * D_FF * 4);
    float*    atomft  = (float*)alloc((size_t)ROWS * D_MODEL * 4);
    int*      segs    = (int*)alloc((size_t)ROWS * 4);
    int*      sege    = (int*)alloc((size_t)ROWS * 4);
    int*      winner  = (int*)alloc((size_t)ROWS * SEG_CAP * 4);
    float*    segbias = (float*)alloc((size_t)ROWS * SEG_CAP * H_HEADS * 4);
    (void)ws_size; (void)in_sizes; (void)n_in; (void)out_size;

    auto cdiv = [](int a, int b) { return (a + b - 1) / b; };

    // ---- weight repack to f16, TRANSPOSED [N][K] (concat along N) ----
    conv_f16t_kernel<<<cdiv(128 * 128, 256), 256, 0, stream>>>(w_q,   WqkvgT, 128, 128, 128, 0);
    conv_f16t_kernel<<<cdiv(128 * 128, 256), 256, 0, stream>>>(w_k,   WqkvgT, 128, 128, 128, 128);
    conv_f16t_kernel<<<cdiv(128 * 128, 256), 256, 0, stream>>>(w_v,   WqkvgT, 128, 128, 128, 256);
    conv_f16t_kernel<<<cdiv(128 * 128, 256), 256, 0, stream>>>(w_g,   WqkvgT, 128, 128, 128, 384);
    conv_f16t_kernel<<<cdiv(128 * 128, 256), 256, 0, stream>>>(w_o,   WoT,    128, 128, 128, 0);
    conv_f16t_kernel<<<cdiv(128 * 512, 256), 256, 0, stream>>>(sw_w1, W12T,   128, 512, 128, 0);
    conv_f16t_kernel<<<cdiv(128 * 512, 256), 256, 0, stream>>>(sw_w2, W12T,   128, 512, 128, 512);
    conv_f16t_kernel<<<cdiv(512 * 128, 256), 256, 0, stream>>>(sw_w3, W3T,    512, 128, 512, 0);
    conv_f16t_kernel<<<cdiv(128 * 512, 256), 256, 0, stream>>>(tok_w, WtokT,  128, 512, 128, 0);

    // ---- LN1 -> f16, then fused QKVG projection (WMMA) ----
    ln1_kernel<<<ROWS / 8, 256, 0, stream>>>(c_atom, ln_a_g, ln_a_b, qn16, ROWS);
    wmma_gemm_kernel<128><<<dim3(ROWS / 64, 512 / 32), 128, 0, stream>>>(
        qn16, WqkvgT, QKVG, nullptr, nullptr, nullptr, ROWS, 512);

    // ---- segment bounds + pair-bias scatter (last-write-wins) ----
    seg_bounds_kernel<<<cdiv(ROWS, 256), 256, 0, stream>>>(tok, segs, sege);
    fill_u32_kernel<<<cdiv(ROWS * SEG_CAP, 256), 256, 0, stream>>>(
        (unsigned*)winner, 0xFFFFFFFFu, ROWS * SEG_CAP);
    fill_u32_kernel<<<cdiv(ROWS * SEG_CAP * H_HEADS, 256), 256, 0, stream>>>(
        (unsigned*)segbias, 0u, ROWS * SEG_CAP * H_HEADS);
    pair_scatter1_kernel<<<cdiv(B_SZ * N_PAIRS, 256), 256, 0, stream>>>(
        p_lm_idx, tok, segs, winner);
    pair_scatter2_kernel<<<cdiv(B_SZ * N_PAIRS, 256), 256, 0, stream>>>(
        p_lm_idx, tok, segs, winner, p_lm, pb_w, pb_b, segbias);

    // ---- segment-local attention (exact fp32) ----
    attn_kernel<<<cdiv(ROWS * H_HEADS, 256), 256, 0, stream>>>(QKVG, segbias, segs, sege, attout);

    // ---- O projection (WMMA), gate+residual+LN2 ----
    wmma_gemm_kernel<128><<<dim3(ROWS / 64, 128 / 32), 128, 0, stream>>>(
        attout, WoT, oproj, nullptr, nullptr, nullptr, ROWS, 128);
    ln2_gate_kernel<<<ROWS / 8, 256, 0, stream>>>(c_atom, QKVG, oproj, ln_f_g, ln_f_b,
                                                  qres, h16, ROWS);

    // ---- SwiGLU FFN (WMMA x2) with fused residual, f16 requant in epilogue ----
    wmma_gemm_kernel<128><<<dim3(ROWS / 64, 1024 / 32), 128, 0, stream>>>(
        h16, W12T, u, nullptr, nullptr, nullptr, ROWS, 1024);
    silu_mul_kernel<<<cdiv(ROWS * D_FF, 256), 256, 0, stream>>>(u, act16);
    wmma_gemm_kernel<512><<<dim3(ROWS / 64, 128 / 32), 128, 0, stream>>>(
        act16, W3T, nullptr, q2_16, qres, nullptr, ROWS, 128);

    // ---- token projection (WMMA, fused bias) ----
    wmma_gemm_kernel<128><<<dim3(ROWS / 64, 512 / 32), 128, 0, stream>>>(
        q2_16, WtokT, atomft, nullptr, nullptr, tok_b, ROWS, 512);

    // ---- segment mean over tokens ----
    token_reduce_kernel<<<B_SZ * N_TOK, 256, 0, stream>>>(atomft, tok, out);
}
